// CrossAttentionFusion_32220844654890
// MI455X (gfx1250) — matrix operand order repaired and checked by reading, not compiled, not run on hardware
//
#include <hip/hip_runtime.h>
#include <hip/hip_bf16.h>

// ---------- types ----------
typedef __attribute__((ext_vector_type(16))) __bf16 v16bf;
typedef __attribute__((ext_vector_type(8)))  float  v8f;

union BF16x16 { v16bf v; unsigned short h[16]; };
union U128    { uint4 u; unsigned short h[8]; };
union FU      { float f; unsigned u; };

static __device__ __forceinline__ unsigned short f2bfu(float f) {
    FU x; x.f = f;
    unsigned r = (x.u + 0x7FFFu + ((x.u >> 16) & 1u)) >> 16;   // RNE
    return (unsigned short)r;
}

#define CDIM   256
#define NPIX   1024   // H*W
#define NB     4
#define NHEADS 8
#define DK     32
#define NKEYS  3072   // 3*NPIX

// ======================================================================
// Kernel 0: s = gamma_s * x_s, fp32 -> bf16, into sbuf[(s*4+b)][C][N]
// ======================================================================
__global__ void scale3_kernel(const float* __restrict__ x0, const float* __restrict__ x1,
                              const float* __restrict__ x2, const float* __restrict__ g0,
                              const float* __restrict__ g1, const float* __restrict__ g2,
                              unsigned short* __restrict__ sbuf) {
    size_t i = (size_t)blockIdx.x * 256 + threadIdx.x;     // [0, 3*1048576)
    int s    = (int)(i >> 20);
    size_t j = i & 1048575u;
    const float* x = (s == 0) ? x0 : ((s == 1) ? x1 : x2);
    float g        = (s == 0) ? g0[0] : ((s == 1) ? g1[0] : g2[0]);
    sbuf[i] = f2bfu(g * x[j]);
}

// ======================================================================
// Kernel 1: projection GEMM  Y[o,n] = sum_c W[o,c]*X[c,n] + b[o]
//   X: bf16 [C, NPIX] chunk selected by blockIdx.y (= s*4+b)
//   MODE 0: bf16 dst[(b*C+o)*Ntot + s*1024 + n]                 (V == V^T/head)
//   MODE 1: bf16 dst[((b*8+o/32)*Ntot + s*1024 + n)*32 + o%32]  (Q, K: [b,h,key,dk])
//   MODE 2: f32  dst[(b*C+o)*1024 + n]                          (final output)
// block = 128 threads (4 waves); grid = (Npix/16, nchunks)
// ======================================================================
template <int MODE>
__global__ void proj_kernel(const unsigned short* __restrict__ Xbase,
                            const float* __restrict__ W,
                            const float* __restrict__ bias,
                            float post_scale, void* __restrict__ dst, int Ntot) {
    __shared__ unsigned short sx[16 * CDIM];   // transposed: sx[n][c], 8 KB

    const int y  = blockIdx.y;
    const int s  = y >> 2;
    const int b  = y & 3;
    const int n0 = blockIdx.x * 16;
    const unsigned short* X = Xbase + (size_t)y * CDIM * NPIX;

    // ---- stage X tile (256 channels x 16 pixels) into LDS, transposed ----
    for (int r = threadIdx.x; r < CDIM; r += 128) {
        const uint4* src = reinterpret_cast<const uint4*>(X + (size_t)r * NPIX + n0);
        U128 c0, c1;
        c0.u = src[0];
        c1.u = src[1];
#pragma unroll
        for (int i = 0; i < 8; ++i) {
            sx[i * CDIM + r]       = c0.h[i];
            sx[(i + 8) * CDIM + r] = c1.h[i];
        }
    }
    __syncthreads();

    const int lane  = threadIdx.x & 31;
    const int wave  = threadIdx.x >> 5;
    const int lrow  = lane & 15;
    const int lhalf = lane >> 4;

    // ---- preload all 8 B-fragments once per wave (reused by 4 output tiles)
    BF16x16 Bf[8];
#pragma unroll
    for (int k = 0; k < 8; ++k) {
        const unsigned short* bp = &sx[lrow * CDIM + k * 32 + lhalf * 16];
        U128 b0, b1;
        b0.u = *reinterpret_cast<const uint4*>(bp);
        b1.u = *reinterpret_cast<const uint4*>(bp + 8);
#pragma unroll
        for (int i = 0; i < 8; ++i) {
            Bf[k].h[i]     = b0.h[i];
            Bf[k].h[8 + i] = b1.h[i];
        }
    }

    for (int ot = wave; ot < 16; ot += 4) {
        const int o0 = ot * 16;
        v8f acc = {};
#pragma unroll
        for (int k = 0; k < 8; ++k) {
            const int c0 = k * 32;
            // A fragment: W rows (fp32 -> bf16). K chunks {lhalf*16, lhalf*16+8}
            const float* wp0 = W + (size_t)(o0 + lrow) * CDIM + c0 + lhalf * 16;
            const float* wp1 = wp0 + 8;
            BF16x16 A;
#pragma unroll
            for (int i = 0; i < 8; ++i) {
                A.h[i]     = f2bfu(wp0[i]);
                A.h[8 + i] = f2bfu(wp1[i]);
            }
            acc = __builtin_amdgcn_wmma_f32_16x16x32_bf16(false, A.v, false, Bf[k].v,
                                                          (short)0, acc, false, false);
        }
        // ---- epilogue + store (D layout: row = o0+v+lhalf*8, col = n0+lrow)
        const int col  = n0 + lrow;
        const int col0 = s * 1024;
#pragma unroll
        for (int v = 0; v < 8; ++v) {
            const int o = o0 + v + lhalf * 8;
            float val   = (acc[v] + bias[o]) * post_scale;
            if (MODE == 2) {
                ((float*)dst)[((size_t)(b * CDIM + o)) * 1024 + col] = val;
            } else if (MODE == 1) {
                ((unsigned short*)dst)[(((size_t)(b * NHEADS + (o >> 5))) * Ntot + col0 + col) * DK +
                                       (o & 31)] = f2bfu(val);
            } else {
                ((unsigned short*)dst)[((size_t)(b * CDIM + o)) * Ntot + col0 + col] = f2bfu(val);
            }
        }
    }
}

// ======================================================================
// Kernel 2: attention. One wave per 16-query tile per (b,h).
//   Q: bf16 [b,h,1024,32] (pre-scaled by 1/sqrt(dk))
//   K: bf16 [b,h,3072,32]
//   V: bf16 [b, C, 3072]  (== V^T per head)
//   O: bf16 [b, C, 1024]
// block = 128 (4 waves); grid = (16, B*NHEADS)
// sP is strictly per-wave; CDNA5 LDS ops from one wave execute in order,
// so a compiler-only wave_barrier suffices (no cross-wave s_barrier).
// ======================================================================
__global__ void attn_kernel(const unsigned short* __restrict__ Qb,
                            const unsigned short* __restrict__ Kb,
                            const unsigned short* __restrict__ Vb,
                            unsigned short* __restrict__ Ob) {
    __shared__ unsigned short sP[4][16 * 32];   // per-wave P tile (16q x 32k), 4 KB

    const int lane  = threadIdx.x & 31;
    const int wave  = threadIdx.x >> 5;
    const int lrow  = lane & 15;
    const int lhalf = lane >> 4;
    const int bh    = blockIdx.y;
    const int b     = bh >> 3;
    const int h     = bh & 7;
    const int qt    = blockIdx.x * 4 + wave;

    // ---- Q A-fragment: two 16B chunks at dk offsets {lhalf*16, lhalf*16+8}
    const unsigned short* Qp = Qb + ((size_t)bh * NPIX + qt * 16 + lrow) * DK;
    U128 q0, q1;
    q0.u = *reinterpret_cast<const uint4*>(Qp + lhalf * 16);
    q1.u = *reinterpret_cast<const uint4*>(Qp + lhalf * 16 + 8);
    BF16x16 Aq;
#pragma unroll
    for (int i = 0; i < 8; ++i) { Aq.h[i] = q0.h[i]; Aq.h[8 + i] = q1.h[i]; }

    const unsigned short* Kbase = Kb + (size_t)bh * NKEYS * DK;

    // ================= pass 1: row max =================
    float m[8];
#pragma unroll
    for (int v = 0; v < 8; ++v) m[v] = -1e30f;

    for (int kt = 0; kt < NKEYS / 16; ++kt) {
        const unsigned short* kp = Kbase + ((size_t)(kt * 16 + lrow)) * DK + lhalf * 16;
        __builtin_prefetch(kp + 16 * DK, 0, 1);
        U128 k0, k1;
        k0.u = *reinterpret_cast<const uint4*>(kp);
        k1.u = *reinterpret_cast<const uint4*>(kp + 8);
        BF16x16 Bk;
#pragma unroll
        for (int i = 0; i < 8; ++i) { Bk.h[i] = k0.h[i]; Bk.h[8 + i] = k1.h[i]; }
        v8f z = {};
        v8f S = __builtin_amdgcn_wmma_f32_16x16x32_bf16(false, Aq.v, false, Bk.v,
                                                        (short)0, z, false, false);
#pragma unroll
        for (int v = 0; v < 8; ++v) m[v] = fmaxf(m[v], S[v]);
    }
#pragma unroll
    for (int off = 1; off < 16; off <<= 1)
#pragma unroll
        for (int v = 0; v < 8; ++v) m[v] = fmaxf(m[v], __shfl_xor(m[v], off, 32));

    // ================= pass 2: exp, P*V =================
    float l[8] = {0, 0, 0, 0, 0, 0, 0, 0};
    v8f O0 = {}, O1 = {};
    const unsigned short* Vbase = Vb + ((size_t)(b * CDIM + h * DK)) * NKEYS;

    for (int kc = 0; kc < NKEYS / 32; ++kc) {
#pragma unroll
        for (int sub = 0; sub < 2; ++sub) {
            const int key0 = kc * 32 + sub * 16;
            const unsigned short* kp = Kbase + ((size_t)(key0 + lrow)) * DK + lhalf * 16;
            U128 k0, k1;
            k0.u = *reinterpret_cast<const uint4*>(kp);
            k1.u = *reinterpret_cast<const uint4*>(kp + 8);
            BF16x16 Bk;
#pragma unroll
            for (int i = 0; i < 8; ++i) { Bk.h[i] = k0.h[i]; Bk.h[8 + i] = k1.h[i]; }
            v8f z = {};
            v8f S = __builtin_amdgcn_wmma_f32_16x16x32_bf16(false, Aq.v, false, Bk.v,
                                                            (short)0, z, false, false);
#pragma unroll
            for (int v = 0; v < 8; ++v) {
                float p = __expf(S[v] - m[v]);
                l[v] += p;
                sP[wave][(v + lhalf * 8) * 32 + sub * 16 + lrow] = f2bfu(p);
            }
        }
        __builtin_amdgcn_wave_barrier();   // per-wave LDS is in-order; block motion only

        // P A-fragment (transposed through LDS)
        const unsigned short* pp = &sP[wave][lrow * 32];
        U128 p0, p1;
        p0.u = *reinterpret_cast<const uint4*>(pp + lhalf * 16);
        p1.u = *reinterpret_cast<const uint4*>(pp + lhalf * 16 + 8);
        BF16x16 Ap;
#pragma unroll
        for (int i = 0; i < 8; ++i) { Ap.h[i] = p0.h[i]; Ap.h[8 + i] = p1.h[i]; }

        // V B-fragments (dk 0..15 and 16..31), contiguous along keys
#pragma unroll
        for (int dh = 0; dh < 2; ++dh) {
            const unsigned short* vp =
                Vbase + (size_t)(dh * 16 + lrow) * NKEYS + kc * 32 + lhalf * 16;
            U128 v0, v1;
            v0.u = *reinterpret_cast<const uint4*>(vp);
            v1.u = *reinterpret_cast<const uint4*>(vp + 8);
            BF16x16 Bv;
#pragma unroll
            for (int i = 0; i < 8; ++i) { Bv.h[i] = v0.h[i]; Bv.h[8 + i] = v1.h[i]; }
            if (dh == 0)
                O0 = __builtin_amdgcn_wmma_f32_16x16x32_bf16(false, Ap.v, false, Bv.v,
                                                             (short)0, O0, false, false);
            else
                O1 = __builtin_amdgcn_wmma_f32_16x16x32_bf16(false, Ap.v, false, Bv.v,
                                                             (short)0, O1, false, false);
        }
        __builtin_amdgcn_wave_barrier();   // keep next chunk's sP writes after reads
    }

    // row sums
#pragma unroll
    for (int off = 1; off < 16; off <<= 1)
#pragma unroll
        for (int v = 0; v < 8; ++v) l[v] += __shfl_xor(l[v], off, 32);

    // store: channel = h*32 + (lrow | +16), column = query index
    unsigned short* Obase = Ob + ((size_t)(b * CDIM + h * DK)) * NPIX;
#pragma unroll
    for (int v = 0; v < 8; ++v) {
        const int q = qt * 16 + v + lhalf * 8;
        const float r = 1.0f / l[v];
        Obase[(size_t)lrow * NPIX + q]        = f2bfu(O0[v] * r);
        Obase[(size_t)(16 + lrow) * NPIX + q] = f2bfu(O1[v] * r);
    }
}

// ======================================================================
// launch
// ======================================================================
extern "C" void kernel_launch(void* const* d_in, const int* in_sizes, int n_in,
                              void* d_out, int out_size, void* d_ws, size_t ws_size,
                              hipStream_t stream) {
    const float* x0 = (const float*)d_in[0];
    const float* x1 = (const float*)d_in[1];
    const float* x2 = (const float*)d_in[2];
    const float* g0 = (const float*)d_in[3];
    const float* g1 = (const float*)d_in[4];
    const float* g2 = (const float*)d_in[5];
    const float* Wq = (const float*)d_in[6];
    const float* bq = (const float*)d_in[7];
    const float* Wk = (const float*)d_in[8];
    const float* bk = (const float*)d_in[9];
    const float* Wv = (const float*)d_in[10];
    const float* bv = (const float*)d_in[11];
    const float* Wo = (const float*)d_in[12];
    const float* bo = (const float*)d_in[13];
    float* out = (float*)d_out;

    unsigned short* sbuf = (unsigned short*)d_ws;                  // 3*4*256*1024
    unsigned short* Qbuf = sbuf + (size_t)3 * NB * CDIM * NPIX;    // 4*8*1024*32
    unsigned short* Kbuf = Qbuf + (size_t)NB * NHEADS * NPIX * DK; // 4*8*3072*32
    unsigned short* Vbuf = Kbuf + (size_t)NB * NHEADS * NKEYS * DK;// 4*256*3072
    unsigned short* Abuf = Vbuf + (size_t)NB * CDIM * NKEYS;       // 4*256*1024

    const float qscale = 0.17677669529663687f;   // 1/sqrt(32), folded into Q

    scale3_kernel<<<dim3((3u * NB * CDIM * NPIX) / 256), 256, 0, stream>>>(
        x0, x1, x2, g0, g1, g2, sbuf);

    // Q (stream 0 only), K & V (3 streams each)
    proj_kernel<1><<<dim3(NPIX / 16, NB), 128, 0, stream>>>(sbuf, Wq, bq, qscale, Qbuf, 1024);
    proj_kernel<1><<<dim3(NPIX / 16, NB * 3), 128, 0, stream>>>(sbuf, Wk, bk, 1.0f, Kbuf, 3072);
    proj_kernel<0><<<dim3(NPIX / 16, NB * 3), 128, 0, stream>>>(sbuf, Wv, bv, 1.0f, Vbuf, 3072);

    attn_kernel<<<dim3(16, NB * NHEADS), 128, 0, stream>>>(Qbuf, Kbuf, Vbuf, Abuf);

    proj_kernel<2><<<dim3(NPIX / 16, NB), 128, 0, stream>>>(Abuf, Wo, bo, 1.0f, out, 1024);
}
